// MultiHeadAttention_18708877541945
// MI455X (gfx1250) — matrix operand order repaired and checked by reading, not compiled, not run on hardware
//
#include <hip/hip_runtime.h>
#include <stdint.h>

// ---------------------------------------------------------------------------
// MI455X (gfx1250) fused MHA: bf16 WMMA GEMMs + flash attention,
// TDM (tensor_load_to_lds) tile staging with LDS padding, double-buffered.
// Shapes: B=2, S=2048, D_IN=D_OUT=1024, H=16, HD=64.  wave32.
// ---------------------------------------------------------------------------

typedef __bf16 bf16;
typedef bf16     v16bf __attribute__((ext_vector_type(16)));
typedef float    v8f   __attribute__((ext_vector_type(8)));
typedef unsigned v4u   __attribute__((ext_vector_type(4)));
typedef int      v8i   __attribute__((ext_vector_type(8)));
typedef int      v4i   __attribute__((ext_vector_type(4)));

union Frag { v16bf v; uint4 q[2]; };

__device__ inline v8f vzero8() {
  v8f z = {0.f, 0.f, 0.f, 0.f, 0.f, 0.f, 0.f, 0.f};
  return z;
}

__device__ inline v8f wmma_bf16(v16bf a, v16bf b, v8f c) {
  // D = A(16x32 bf16) * B(32x16 bf16) + C(16x16 f32)
  return __builtin_amdgcn_wmma_f32_16x16x32_bf16(
      false, a, false, b, (short)0, c, false, false);
}

// ---------------------------------------------------------------------------
// TDM: 2D tile load Global->LDS, bf16 elements, optional LDS row padding.
// Descriptor bitfields per CDNA5 ISA.
//   pad_interval_code: pad after 2^(code+1) DWORDs; pad_amount_dw DWORDs pad.
// This toolchain exposes the 6-arg builtin (groups 2/3 + extension group).
// ---------------------------------------------------------------------------
__device__ inline void tdm_load_2d_bf16(uint32_t lds_off, const void* gptr,
                                        uint32_t tensor_d0, uint32_t tensor_d1,
                                        uint32_t tile_d0, uint32_t tile_d1,
                                        uint32_t stride0,
                                        uint32_t pad_interval_code,
                                        uint32_t pad_amount_dw) {
  uint64_t ga = (uint64_t)(uintptr_t)gptr;
  v4u g0;
  g0.x = 1u;                                                  // count=1, user D#
  g0.y = lds_off;                                             // lds_addr
  g0.z = (uint32_t)(ga & 0xFFFFFFFFu);                        // global_addr lo
  g0.w = (uint32_t)((ga >> 32) & 0x01FFFFFFu) | (2u << 30);   // addr hi | type=2
  v8i g1;
  uint32_t w0 = (1u << 16)                                    // data_size = 2B
              | (1u << 20)                                    // pad_enable
              | (pad_interval_code << 22)
              | ((pad_amount_dw - 1u) << 25);
  g1[0] = (int)w0;
  g1[1] = (int)(tensor_d0 << 16);                             // dim0[15:0] @63:48
  g1[2] = (int)((tensor_d0 >> 16) | (tensor_d1 << 16));       // dim0 hi | dim1 lo
  g1[3] = (int)((tensor_d1 >> 16) | (tile_d0 << 16));         // dim1 hi | tile0
  g1[4] = (int)(tile_d1 & 0xFFFFu);                           // tile1 | tile2=0
  g1[5] = (int)stride0;                                       // dim0_stride lo32
  g1[6] = 0;                                                  // stride0 hi | stride1 lo
  g1[7] = 0;
  v4i z4 = {0, 0, 0, 0};
  v8i z8 = {0, 0, 0, 0, 0, 0, 0, 0};
  __builtin_amdgcn_tensor_load_to_lds(g0, g1, z4, z4, z8, 0);
}

// ---------------------------------------------------------------------------
// fp32 -> bf16 conversion (row-major copy)
// ---------------------------------------------------------------------------
__global__ __launch_bounds__(256) void cvt_f32_bf16(const float* __restrict__ src,
                                                    bf16* __restrict__ dst, int n) {
  int i = blockIdx.x * 1024 + threadIdx.x * 4;
  if (i + 3 < n) {
    float4 f = *(const float4*)(src + i);
    alignas(8) bf16 t[4];
    t[0] = (bf16)f.x; t[1] = (bf16)f.y; t[2] = (bf16)f.z; t[3] = (bf16)f.w;
    *(uint2*)(dst + i) = *(const uint2*)t;
  }
}

// fp32 [K][N] -> bf16 transposed [N][K], 32x32 LDS tiles, coalesced both sides.
__global__ __launch_bounds__(256) void cvt_transpose_bf16(
    const float* __restrict__ src, bf16* __restrict__ dst, int K, int N) {
  __shared__ bf16 t[32][33];
  const int bn = blockIdx.x * 32, bk = blockIdx.y * 32;
  const int tx = threadIdx.x & 31, ty = threadIdx.x >> 5;  // ty 0..7
#pragma unroll
  for (int j = 0; j < 32; j += 8)
    t[ty + j][tx] = (bf16)src[(size_t)(bk + ty + j) * N + bn + tx];
  __syncthreads();
#pragma unroll
  for (int j = 0; j < 32; j += 8)
    dst[(size_t)(bn + ty + j) * K + bk + tx] = t[tx][ty + j];
}

// ---------------------------------------------------------------------------
// Tiled bf16 GEMM: C[M,N] = A[M,K] * Wt[N,K]^T, fp32 accumulate.
// 256 thr = 8 waves, 128x128x32 tile, wave grid 2(M) x 4(N), 8 WMMA/wave/iter.
// Both A and B tiles staged by TDM with a 4-DWORD pad per 16-DWORD row
// (row stride 40 bf16), double-buffered: wave 0 issues next tile's two
// descriptors, waits tensorcnt<=2 for the current pair, block barrier.
// ---------------------------------------------------------------------------
template <bool OUT_BF16>
__global__ __launch_bounds__(256) void gemm_bf16_kernel(
    const bf16* __restrict__ A, const bf16* __restrict__ Wt,
    void* __restrict__ Cp, int M, int N, int K) {
  __shared__ bf16 As[2][128][40];   // [buf][m][k], pad 8
  __shared__ bf16 Bs[2][128][40];   // [buf][n][k], pad 8

  const int tid  = threadIdx.x;
  const int lane = tid & 31;
  const int w    = tid >> 5;
  const int wm   = w >> 2;          // 0..1
  const int wn   = w & 3;           // 0..3
  const int l15  = lane & 15;
  const int half = lane >> 4;

  const int bm = blockIdx.y * 128;
  const int bn = blockIdx.x * 128;

  v8f acc[4][2];
#pragma unroll
  for (int mt = 0; mt < 4; ++mt)
#pragma unroll
    for (int nt = 0; nt < 2; ++nt) acc[mt][nt] = vzero8();

  const uint32_t asOff[2] = {(uint32_t)(uintptr_t)&As[0][0][0],
                             (uint32_t)(uintptr_t)&As[1][0][0]};
  const uint32_t bsOff[2] = {(uint32_t)(uintptr_t)&Bs[0][0][0],
                             (uint32_t)(uintptr_t)&Bs[1][0][0]};

  // prologue: stage k0=0 into buffer 0
  if (w == 0) {
    tdm_load_2d_bf16(asOff[0], A  + (size_t)bm * K, K, M, 32, 128, K, 3, 4);
    tdm_load_2d_bf16(bsOff[0], Wt + (size_t)bn * K, K, N, 32, 128, K, 3, 4);
  }

  int cur = 0;
  for (int k0 = 0; k0 < K; k0 += 32, cur ^= 1) {
    if (w == 0) {
      if (k0 + 32 < K) {
        tdm_load_2d_bf16(asOff[cur ^ 1], A  + (size_t)bm * K + (k0 + 32),
                         K, M, 32, 128, K, 3, 4);
        tdm_load_2d_bf16(bsOff[cur ^ 1], Wt + (size_t)bn * K + (k0 + 32),
                         K, N, 32, 128, K, 3, 4);
        __builtin_amdgcn_s_wait_tensorcnt(2);   // current pair done (in-order)
      } else {
        __builtin_amdgcn_s_wait_tensorcnt(0);
      }
    }
    __syncthreads();

    Frag a[4], b[2];
#pragma unroll
    for (int mt = 0; mt < 4; ++mt) {
      const bf16* base = &As[cur][wm * 64 + mt * 16 + l15][0];
      a[mt].q[0] = *(const uint4*)(base + half * 8);       // K = khi*8 + 0..7
      a[mt].q[1] = *(const uint4*)(base + 16 + half * 8);  // K = 16+khi*8+0..7
    }
#pragma unroll
    for (int nt = 0; nt < 2; ++nt) {
      const bf16* base = &Bs[cur][wn * 32 + nt * 16 + l15][0];
      b[nt].q[0] = *(const uint4*)(base + half * 8);       // K = half*8 + 0..7
      b[nt].q[1] = *(const uint4*)(base + 16 + half * 8);  // K = 16+half*8+0..7
    }
#pragma unroll
    for (int mt = 0; mt < 4; ++mt)
#pragma unroll
      for (int nt = 0; nt < 2; ++nt)
        acc[mt][nt] = wmma_bf16(a[mt].v, b[nt].v, acc[mt][nt]);

    __syncthreads();   // all waves done with buf `cur` before TDM overwrites it
  }

  // C layout: VGPR r -> row r+8*half, col = lane&15
#pragma unroll
  for (int mt = 0; mt < 4; ++mt)
#pragma unroll
    for (int nt = 0; nt < 2; ++nt)
#pragma unroll
      for (int r = 0; r < 8; ++r) {
        int row = bm + wm * 64 + mt * 16 + r + 8 * half;
        int col = bn + wn * 32 + nt * 16 + l15;
        float v = acc[mt][nt][r];
        if (OUT_BF16) ((bf16*)Cp)[(size_t)row * N + col] = (bf16)v;
        else          ((float*)Cp)[(size_t)row * N + col] = v;
      }
}

// ---------------------------------------------------------------------------
// Flash attention. Heads are contiguous [S,64] slabs at (b*H+h)*S*64.
// One block per (bh, 128 q rows), 8 waves x 16 q rows, k tiles of 32 keys.
// K tile staged by TDM (pad 4 DW per 32-DW row -> stride 72); V staged
// transposed manually; row-sum of P computed by an extra WMMA vs all-ones.
// ---------------------------------------------------------------------------
__global__ __launch_bounds__(256) void attn_kernel(
    const bf16* __restrict__ Q, const bf16* __restrict__ Kx,
    const bf16* __restrict__ V, bf16* __restrict__ O) {
  __shared__ bf16 Ks[32][72];      // [key][d]   (TDM, natural + pad)
  __shared__ bf16 Vt[64][40];      // [d][key]   (manual transpose)
  __shared__ bf16 Pl[8][16][40];   // per-wave P staging [qrow][key]

  const int tid  = threadIdx.x;
  const int lane = tid & 31;
  const int w    = tid >> 5;
  const int l15  = lane & 15;
  const int half = lane >> 4;

  const int bid   = blockIdx.x;
  const int qblk  = bid & 15;       // S/128 = 16
  const int bh    = bid >> 4;       // b*H + h
  const size_t hb = (size_t)bh * (2048u * 64u);

  const bf16* Qh = Q + hb;
  const bf16* Kh = Kx + hb;
  const bf16* Vh = V + hb;
  bf16*       Oh = O + hb;

  const int q0 = qblk * 128;
  const int qb = q0 + w * 16;

  // Q fragments (A layout), constant across the k loop.
  Frag qa[2];
  {
    const bf16* qrow = Qh + (size_t)(qb + l15) * 64;
    qa[0].q[0] = *(const uint4*)(qrow + half * 8);
    qa[0].q[1] = *(const uint4*)(qrow + 16 + half * 8);
    qa[1].q[0] = *(const uint4*)(qrow + 32 + half * 8);
    qa[1].q[1] = *(const uint4*)(qrow + 48 + half * 8);
  }

  // all-ones B fragment for row-sum WMMA (bf16 1.0 = 0x3F80)
  Frag ones;
  {
    uint4 u = {0x3F803F80u, 0x3F803F80u, 0x3F803F80u, 0x3F803F80u};
    ones.q[0] = u; ones.q[1] = u;
  }

  const uint32_t ksOff = (uint32_t)(uintptr_t)&Ks[0][0];

  v8f o[4];
  float mrow[8], lrow[8];
#pragma unroll
  for (int nt = 0; nt < 4; ++nt) o[nt] = vzero8();
#pragma unroll
  for (int r = 0; r < 8; ++r) { mrow[r] = -3.0e38f; lrow[r] = 0.f; }

  const float scale = 0.125f;   // 1/sqrt(64)
  const int kend = q0 + 128;

  for (int kb = 0; kb < kend; kb += 32) {
    // ---- K tile via TDM: 32x64 bf16 rows, +16B pad/row ----
    if (w == 0)
      tdm_load_2d_bf16(ksOff, Kh + (size_t)kb * 64, 64, 2048, 64, 32, 64, 4, 4);
    // ---- V tile transposed, cooperative ----
    {
      int r = tid >> 3, c0 = (tid & 7) * 8;
      uint4 u = *(const uint4*)(Vh + (size_t)(kb + r) * 64 + c0);
      alignas(16) bf16 tmp[8];
      *(uint4*)tmp = u;
#pragma unroll
      for (int j = 0; j < 8; ++j) Vt[c0 + j][r] = tmp[j];
    }
    if (w == 0) __builtin_amdgcn_s_wait_tensorcnt(0);
    __syncthreads();

    // ---- scores: S(16q x 32k) = Q(16x64) * K^T ----
    v8f s0 = vzero8(), s1 = vzero8();
#pragma unroll
    for (int d0 = 0; d0 < 2; ++d0) {
      Frag b0, b1;
      const bf16* p0 = &Ks[l15][d0 * 32 + half * 8];
      b0.q[0] = *(const uint4*)p0;       b0.q[1] = *(const uint4*)(p0 + 16);
      const bf16* p1 = &Ks[16 + l15][d0 * 32 + half * 8];
      b1.q[0] = *(const uint4*)p1;       b1.q[1] = *(const uint4*)(p1 + 16);
      s0 = wmma_bf16(qa[d0].v, b0.v, s0);
      s1 = wmma_bf16(qa[d0].v, b1.v, s1);
    }

    // ---- causal mask + running max (row m = r + 8*half) ----
    float alpha[8];
#pragma unroll
    for (int r = 0; r < 8; ++r) {
      int qi = qb + r + 8 * half;
      float v0 = s0[r] * scale, v1 = s1[r] * scale;
      if (kb + l15 > qi)      v0 = -3.0e38f;
      if (kb + 16 + l15 > qi) v1 = -3.0e38f;
      float mt = fmaxf(v0, v1);
#pragma unroll
      for (int off = 8; off >= 1; off >>= 1)
        mt = fmaxf(mt, __shfl_xor(mt, off, 32));
      float mnew = fmaxf(mrow[r], mt);
      float a  = __expf(mrow[r] - mnew);
      float p0 = __expf(v0 - mnew);
      float p1 = __expf(v1 - mnew);
      mrow[r]  = mnew;
      alpha[r] = a;
      Pl[w][r + 8 * half][l15]      = (bf16)p0;   // C-layout -> LDS
      Pl[w][r + 8 * half][16 + l15] = (bf16)p1;
    }

    // rescale running output
#pragma unroll
    for (int nt = 0; nt < 4; ++nt)
#pragma unroll
      for (int r = 0; r < 8; ++r) o[nt][r] *= alpha[r];

    // wave-private LDS round-trip: enforce store->load order across lanes
    asm volatile("s_wait_dscnt 0" ::: "memory");

    // ---- P fragment (A layout, K = 32 keys) ----
    Frag pa;
    {
      const bf16* prow = &Pl[w][l15][0];
      pa.q[0] = *(const uint4*)(prow + half * 8);
      pa.q[1] = *(const uint4*)(prow + 16 + half * 8);
    }

    // ---- row-sum of P via WMMA against all-ones (every col = rowsum) ----
    v8f ls = wmma_bf16(pa.v, ones.v, vzero8());
#pragma unroll
    for (int r = 0; r < 8; ++r) lrow[r] = lrow[r] * alpha[r] + ls[r];

    // ---- O(16x64) += P(16x32) * V(32x64) ----
#pragma unroll
    for (int nt = 0; nt < 4; ++nt) {
      Frag vb;
      const bf16* vcol = &Vt[nt * 16 + l15][half * 8];
      vb.q[0] = *(const uint4*)vcol;
      vb.q[1] = *(const uint4*)(vcol + 16);
      o[nt] = wmma_bf16(pa.v, vb.v, o[nt]);
    }

    __syncthreads();
  }

  // ---- normalize and store context ----
#pragma unroll
  for (int r = 0; r < 8; ++r) {
    float inv = 1.0f / lrow[r];
    bf16* orow = Oh + (size_t)(qb + r + 8 * half) * 64;
#pragma unroll
    for (int nt = 0; nt < 4; ++nt)
      orow[nt * 16 + l15] = (bf16)(o[nt][r] * inv);
  }
}

// ---------------------------------------------------------------------------
// Launch
// ---------------------------------------------------------------------------
extern "C" void kernel_launch(void* const* d_in, const int* in_sizes, int n_in,
                              void* d_out, int out_size, void* d_ws, size_t ws_size,
                              hipStream_t stream) {
  const float* x  = (const float*)d_in[0];
  const float* Wq = (const float*)d_in[1];
  const float* Wk = (const float*)d_in[2];
  const float* Wv = (const float*)d_in[3];
  const float* Wp = (const float*)d_in[4];
  float* out = (float*)d_out;

  const int S = 2048, D = 1024, M = 2 * S;  // M = B*S = 4096

  char* ws = (char*)d_ws;
  bf16* xb  = (bf16*)ws; ws += (size_t)M * D * 2;   // reused as ctx buffer
  bf16* wqb = (bf16*)ws; ws += (size_t)D * D * 2;   // transposed [N][K]
  bf16* wkb = (bf16*)ws; ws += (size_t)D * D * 2;
  bf16* wvb = (bf16*)ws; ws += (size_t)D * D * 2;
  bf16* wpb = (bf16*)ws; ws += (size_t)D * D * 2;
  bf16* Qb  = (bf16*)ws; ws += (size_t)M * D * 2;
  bf16* Kb  = (bf16*)ws; ws += (size_t)M * D * 2;
  bf16* Vb  = (bf16*)ws; ws += (size_t)M * D * 2;
  bf16* Cb  = xb;  // x no longer needed after QKV GEMMs (stream-ordered)

  cvt_f32_bf16<<<M * D / 1024, 256, 0, stream>>>(x, xb, M * D);
  dim3 tg(D / 32, D / 32);
  cvt_transpose_bf16<<<tg, 256, 0, stream>>>(Wq, wqb, D, D);
  cvt_transpose_bf16<<<tg, 256, 0, stream>>>(Wk, wkb, D, D);
  cvt_transpose_bf16<<<tg, 256, 0, stream>>>(Wv, wvb, D, D);
  cvt_transpose_bf16<<<tg, 256, 0, stream>>>(Wp, wpb, D, D);

  dim3 g(D / 128, M / 128);
  gemm_bf16_kernel<true><<<g, 256, 0, stream>>>(xb, wqb, Qb, M, D, D);
  gemm_bf16_kernel<true><<<g, 256, 0, stream>>>(xb, wkb, Kb, M, D, D);
  gemm_bf16_kernel<true><<<g, 256, 0, stream>>>(xb, wvb, Vb, M, D, D);

  attn_kernel<<<32 * (S / 128), 256, 0, stream>>>(Qb, Kb, Vb, Cb);

  gemm_bf16_kernel<false><<<g, 256, 0, stream>>>(Cb, wpb, out, M, D, D);
}
